// Encoder_6957847019676
// MI455X (gfx1250) — matrix operand order repaired and checked by reading, not compiled, not run on hardware
//
#include <hip/hip_runtime.h>

typedef __bf16 bf16;
typedef __attribute__((ext_vector_type(16))) __bf16 v16bf;
typedef __attribute__((ext_vector_type(8)))  __bf16 v8bf;
typedef __attribute__((ext_vector_type(4)))  __bf16 v4bf;
typedef __attribute__((ext_vector_type(8)))  float  v8f;
typedef __attribute__((ext_vector_type(4)))  float  v4f;

#define NROWS  16384
#define INDIM  2048
#define OUTDIM 512

#define BM 256
#define BN 128
#define BK 32
#define LDSW 40    // LDS row stride (bf16 elems): 80 B = mult of 16 B, spreads banks
#define CTW  264   // epilogue transpose stride (bf16 elems): 528 B = mult of 16 B
#define SMEM_BYTES 67584   // max(tiles: 2*(BM+BN)*LDSW*2 = 61440, Ct: 128*CTW*2 = 67584)

// ---------------------------------------------------------------------------
// W [K][N] f32  ->  W^T [N][K] bf16   (coalesced read; tiny one-time kernel)
// ---------------------------------------------------------------------------
__global__ void wt_bf16_kernel(const float* __restrict__ w, bf16* __restrict__ wT) {
  int i = blockIdx.x * blockDim.x + threadIdx.x;
  if (i < INDIM * OUTDIM) {
    int k = i / OUTDIM, n = i % OUTDIM;
    wT[(size_t)n * INDIM + k] = (bf16)w[i];
  }
}

// ---------------------------------------------------------------------------
// C = A(f32, MxK row-major, cvt->bf16 during LDS staging)
//       @ B (given as B^T: bf16, [N][K] row-major, ldb = K)
// f32 accumulate via v_wmma_f32_16x16x32_bf16.
// B tiles staged with GLOBAL_LOAD_ASYNC_TO_LDS_B128 (ASYNCcnt), no VGPR traffic.
// Block: 256 threads = 8 wave32 -> 256x128 C tile; waves 4(M) x 2(N),
// each wave owns a 64x64 tile = 4x4 WMMA fragments (16 WMMAs per K-tile).
// OUT_TBF16: write C transposed as bf16 [N][M] (via LDS transpose) for chaining;
// else write f32 row-major with nontemporal stores.
// ---------------------------------------------------------------------------
template <bool OUT_TBF16>
__global__ __launch_bounds__(256)
void gemm_f32a_bf16bt(const float* __restrict__ A, const bf16* __restrict__ Bt,
                      void* __restrict__ Cout, int K, int lda, int ldb, int ldc) {
  __shared__ __align__(16) unsigned char smem[SMEM_BYTES];
  bf16* AsB = (bf16*)smem;             // [2][BM*LDSW]  A tiles (bf16, [m][k])
  bf16* BsB = AsB + 2 * BM * LDSW;     // [2][BN*LDSW]  B tiles (bf16, [n][k])

  const int tid  = threadIdx.x;
  const int lane = tid & 31;
  const int wave = tid >> 5;
  const int l16  = lane & 15;
  const int lh   = lane >> 4;

  const int mBlk = blockIdx.y * BM;
  const int nBlk = blockIdx.x * BN;
  const int wm   = (wave & 3) * 64;    // wave M offset (4 frags)
  const int wn   = (wave >> 2) * 64;   // wave N offset (4 frags)

  v8f acc[4][4];
#pragma unroll
  for (int i = 0; i < 4; ++i)
#pragma unroll
    for (int j = 0; j < 4; ++j)
#pragma unroll
      for (int r = 0; r < 8; ++r) acc[i][j][r] = 0.0f;

  // A staging: 8 passes, row = aRow+32p, 4 floats at aCol
  const int aRow = tid >> 3;           // 0..31
  const int aCol = (tid & 7) * 4;      // 0..28
  // B staging: 2 async b128 per thread: row n = tid/2, k-chunk = (tid&1)*8 (+16)
  const int bN_  = tid >> 1;           // 0..127
  const int bK8  = (tid & 1) * 8;      // 0 or 8

  v4f aReg[8];
  const int KT = K / BK;

  auto stageB = [&](int kt, int buf) {
#pragma unroll
    for (int p = 0; p < 2; ++p) {
      const int k8 = bK8 + 16 * p;                     // 0,8 / 16,24
      unsigned lds = (unsigned)(size_t)&BsB[buf * BN * LDSW + bN_ * LDSW + k8];
      unsigned long long ga = (unsigned long long)(size_t)(
          Bt + (size_t)(nBlk + bN_) * ldb + (size_t)kt * BK + k8);
      // async DMA: 16 B global -> LDS, tracked by ASYNCcnt (no VGPR data path)
      asm volatile("global_load_async_to_lds_b128 %0, %1, off"
                   :: "v"(lds), "v"(ga) : "memory");
    }
  };

  auto stageGA = [&](int kt) {
#pragma unroll
    for (int p = 0; p < 8; ++p)
      aReg[p] = __builtin_nontemporal_load(
          (const v4f*)(A + (size_t)(mBlk + aRow + 32 * p) * lda +
                       (size_t)kt * BK + aCol));
  };

  auto stageLA = [&](int buf) {
#pragma unroll
    for (int p = 0; p < 8; ++p)
      *(v4bf*)&AsB[buf * BM * LDSW + (aRow + 32 * p) * LDSW + aCol] =
          __builtin_convertvector(aReg[p], v4bf);      // v_cvt_pk_bf16_f32 pairs
  };

  auto computeT = [&](int buf) {
    const bf16* As = &AsB[buf * BM * LDSW];
    const bf16* Bs = &BsB[buf * BN * LDSW];
    // A 16x32 frag: lane<16 -> K 0..7 & 16..23 ; lane>=16 -> K 8..15 & 24..31
    v16bf af[4];
#pragma unroll
    for (int i = 0; i < 4; ++i) {
      const bf16* base = As + (wm + 16 * i + l16) * LDSW + lh * 8;
      v8bf lo = *(const v8bf*)(base);
      v8bf hi = *(const v8bf*)(base + 16);
      af[i] = __builtin_shufflevector(lo, hi, 0, 1, 2, 3, 4, 5, 6, 7,
                                      8, 9, 10, 11, 12, 13, 14, 15);
    }
    // B 32x16 frag: lane holds col n = lane%16; lane<16 -> K 0..7 & 16..23 ; lane>=16 -> K 8..15 & 24..31
    v16bf bfr[4];
#pragma unroll
    for (int j = 0; j < 4; ++j) {
      const bf16* base = Bs + (wn + 16 * j + l16) * LDSW + lh * 8;
      v8bf lo = *(const v8bf*)(base);
      v8bf hi = *(const v8bf*)(base + 16);
      bfr[j] = __builtin_shufflevector(lo, hi, 0, 1, 2, 3, 4, 5, 6, 7,
                                       8, 9, 10, 11, 12, 13, 14, 15);
    }
#pragma unroll
    for (int i = 0; i < 4; ++i)
#pragma unroll
      for (int j = 0; j < 4; ++j)
        acc[i][j] = __builtin_amdgcn_wmma_f32_16x16x32_bf16(
            false, af[i], false, bfr[j], (short)0, acc[i][j], false, false);
  };

  // prologue
  stageB(0, 0);
  stageGA(0);
  stageLA(0);
  asm volatile("s_wait_asynccnt 0x0" ::: "memory");
  __syncthreads();

  for (int kt = 0; kt < KT; ++kt) {
    if (kt + 1 < KT) {
      stageB(kt + 1, (kt + 1) & 1);    // async copy overlaps compute
      stageGA(kt + 1);                 // global A loads overlap compute
    }
    if (kt + 2 < KT)                   // streaming prefetch for A
      __builtin_prefetch(A + (size_t)(mBlk + aRow) * lda +
                             (size_t)(kt + 2) * BK + aCol, 0, 0);
    computeT(kt & 1);
    if (kt + 1 < KT) stageLA((kt + 1) & 1);   // cvt + LDS store (other buffer)
    asm volatile("s_wait_asynccnt 0x0" ::: "memory");
    __syncthreads();
  }

  // ------------------------------- epilogue --------------------------------
  // C/D layout: VGPR r -> M = base + r + 8*(lane/16), N = base + lane%16
  if constexpr (OUT_TBF16) {
    // transpose through LDS (reusing tile buffers), then contiguous bf16 rows
    bf16* Ct = (bf16*)smem;            // [128 cols][CTW] over 256 rows
#pragma unroll
    for (int i = 0; i < 4; ++i)
#pragma unroll
      for (int j = 0; j < 4; ++j) {
        const int rowL = wm + 16 * i + lh * 8;
        const int colL = wn + 16 * j + l16;
#pragma unroll
        for (int r = 0; r < 8; ++r)
          Ct[colL * CTW + rowL + r] = (bf16)acc[i][j][r];
      }
    __syncthreads();
    bf16* outp = (bf16*)Cout;
    const int c    = tid >> 1;          // output row (= column of C): 0..127
    const int half = (tid & 1) * 128;
#pragma unroll
    for (int e = 0; e < 16; ++e) {
      v8bf v = *(const v8bf*)&Ct[c * CTW + half + e * 8];
      *(v8bf*)&outp[(size_t)(nBlk + c) * ldc + mBlk + half + e * 8] = v;
    }
  } else {
    float* outp = (float*)Cout;
#pragma unroll
    for (int i = 0; i < 4; ++i)
#pragma unroll
      for (int j = 0; j < 4; ++j) {
        const int row0 = mBlk + wm + 16 * i + lh * 8;
        const int col  = nBlk + wn + 16 * j + l16;
#pragma unroll
        for (int r = 0; r < 8; ++r)
          __builtin_nontemporal_store(acc[i][j][r],
                                      &outp[(size_t)(row0 + r) * ldc + col]);
      }
  }
}

// ---------------------------------------------------------------------------
extern "C" void kernel_launch(void* const* d_in, const int* in_sizes, int n_in,
                              void* d_out, int out_size, void* d_ws,
                              size_t ws_size, hipStream_t stream) {
  const float* x   = (const float*)d_in[0];  // [16384, 2048]
  const float* adj = (const float*)d_in[1];  // [16384, 16384]
  const float* w   = (const float*)d_in[2];  // [2048, 512]
  float* out = (float*)d_out;                // [16384, 512]

  bf16* wT = (bf16*)d_ws;                    // [512][2048]  bf16 (2 MB)
  bf16* hT = wT + (size_t)INDIM * OUTDIM;    // [512][16384] bf16 (16 MB, L2-hot)

  // 1) W -> W^T bf16
  wt_bf16_kernel<<<(INDIM * OUTDIM + 255) / 256, 256, 0, stream>>>(w, wT);

  dim3 block(256);
  dim3 grid(OUTDIM / BN, NROWS / BM);  // (4, 64); same-M blocks adjacent ->
                                       // adj row tiles shared via L2

  // 2) h^T = (x @ W)^T : A = x (f32, NT-streamed), B^T = W^T -> bf16 h^T
  gemm_f32a_bf16bt<true><<<grid, block, 0, stream>>>(
      x, wT, hT, /*K=*/INDIM, /*lda=*/INDIM, /*ldb=*/INDIM, /*ldc=*/NROWS);

  // 3) out = adj @ h : A = adj (f32, streamed once, NT), B^T = h^T (L2)
  gemm_f32a_bf16bt<false><<<grid, block, 0, stream>>>(
      adj, hT, out, /*K=*/NROWS, /*lda=*/NROWS, /*ldb=*/NROWS, /*ldc=*/OUTDIM);
}
